// GraphCapsuleLayer_56684978372667
// MI455X (gfx1250) — matrix operand order repaired and checked by reading, not compiled, not run on hardware
//
#include <hip/hip_runtime.h>
#include <math.h>

#define NTOT 4096
#define DIM  128

typedef __attribute__((ext_vector_type(16))) __bf16          v16bf;
typedef __attribute__((ext_vector_type(8)))  float           v8f;
typedef __attribute__((ext_vector_type(8)))  unsigned short  u16x8;
typedef __attribute__((ext_vector_type(16))) unsigned short  u16x16;
typedef __attribute__((ext_vector_type(4)))  unsigned int    u32x4;
typedef __attribute__((ext_vector_type(8)))  int             i32x8;
typedef __attribute__((ext_vector_type(4)))  int             i32x4;

__device__ __forceinline__ unsigned short f32_to_bf16(float f) {
    union { float f; unsigned int u; } c; c.f = f;
    unsigned int u = c.u;
    unsigned int r = 0x7FFFu + ((u >> 16) & 1u);   // round-to-nearest-even
    return (unsigned short)((u + r) >> 16);
}

// ---------------------------------------------------------------------------
// TDM: async-load a 2D bf16 tile (tile_rows x tile_k elements) from a
// row-major tensor (row length tensor_k, row stride row_stride elements,
// tensor_rows rows) into LDS at byte offset lds_byte_off.
// Descriptor bitfields per CDNA5 ISA Sec. 8.3/8.4.
// clang-23 6-arg builtin form: (g0, g1, g2, g3, g4, cpol).
// ---------------------------------------------------------------------------
__device__ __forceinline__ void tdm_load_tile_bf16(
    const unsigned short* gsrc, unsigned lds_byte_off,
    unsigned tile_k, unsigned tile_rows,
    unsigned tensor_k, unsigned tensor_rows, unsigned row_stride)
{
    unsigned long long ga = (unsigned long long)gsrc;
    u32x4 g0;
    g0[0] = 1u;                                            // count=1, user mode
    g0[1] = lds_byte_off;                                  // lds_addr (bytes)
    g0[2] = (unsigned)(ga & 0xFFFFFFFFu);                  // global_addr[31:0]
    g0[3] = (unsigned)((ga >> 32) & 0x1FFFFFFu)            // global_addr[56:32]
          | (2u << 30);                                    // type = 2 ("image")
    i32x8 g1;
    g1[0] = (int)(1u << 16);                               // data_size=1 -> 2B
    g1[1] = (int)((tensor_k & 0xFFFFu) << 16);             // tensor_dim0[15:0]
    g1[2] = (int)(((tensor_k >> 16) & 0xFFFFu)             // tensor_dim0[31:16]
          | ((tensor_rows & 0xFFFFu) << 16));              // tensor_dim1[15:0]
    g1[3] = (int)(((tensor_rows >> 16) & 0xFFFFu)          // tensor_dim1[31:16]
          | ((tile_k & 0xFFFFu) << 16));                   // tile_dim0
    g1[4] = (int)(tile_rows & 0xFFFFu);                    // tile_dim1 (dim2=0)
    g1[5] = (int)row_stride;                               // dim0_stride[31:0]
    g1[6] = 0;                                             // stride[47:32], d1s
    g1[7] = 0;
    i32x4 gz4 = {0, 0, 0, 0};                              // groups 2/3 unused (2D)
    i32x8 gz8 = {0, 0, 0, 0, 0, 0, 0, 0};                  // trailing group unused
    __builtin_amdgcn_tensor_load_to_lds(g0, g1, gz4, gz4, gz8, 0);
}

// ---------------------------------------------------------------------------
// WMMA GEMM:  C[M x N] (f32) = A[M x K] (bf16 row-major) @ B[K x N]
// B supplied transposed: BT[N x K] bf16 row-major.
// One wave per 32x32 output tile: 4 accumulators, 2 A-frags x 2 B-frags
// (each operand fragment feeds two v_wmma -> 1 load/WMMA).
// B panel (32 rows x CHUNK) staged into *dynamic* LDS by the Tensor Data
// Mover, double-buffered, synchronized with s_wait_tensorcnt (one wave per
// block, TDM ops of a wave complete in order -> no barrier needed).
// Dynamic LDS (external symbol) + the asm memory barrier keep the optimizer
// from folding the staged reads away (the TDM intrinsic's LDS write is
// invisible to alias analysis; an internal, never-stored LDS array would be
// undef-folded).
// K multiple of CHUNK; CHUNK multiple of 32, <= 256. M,N multiples of 32.
// accumulate=1: init accumulators from C (wave owns its tile; no races).
// Launch with 32768 bytes of dynamic LDS.
// ---------------------------------------------------------------------------
__global__ void __launch_bounds__(32)
wmma_gemm_bf16(const unsigned short* __restrict__ A,
               const unsigned short* __restrict__ BT,
               float* __restrict__ C,
               int M, int N, int K, int accumulate, int CHUNK)
{
    extern __shared__ unsigned short Bs[];        // 2 x 16KB double buffer

    const int lane   = threadIdx.x;               // 0..31, one wave
    const int tilesN = N >> 5;
    const int tm     = blockIdx.x / tilesN;
    const int tn     = blockIdx.x % tilesN;
    const int m0     = tm * 32;
    const int n0     = tn * 32;

    const int mLane  = lane & 15;
    const int hiHalf = lane >> 4;                 // 0 or 1
    const int kbaseA = hiHalf ? 16 : 0;           // A-frag K offset per half
    const int koffB  = hiHalf ? 16 : 0;           // B-frag K offset per half

    const size_t aRow0 = (size_t)(m0 + mLane)      * (size_t)K;
    const size_t aRow1 = (size_t)(m0 + 16 + mLane) * (size_t)K;
    const int    bsRow0 =  mLane       * CHUNK;   // LDS row (elements)
    const int    bsRow1 = (16 + mLane) * CHUNK;

    const int rBase = m0 + (hiHalf ? 8 : 0);      // C row base for v8f elem r
    const int cCol  = n0 + mLane;

    v8f acc00, acc01, acc10, acc11;
    if (accumulate) {
        #pragma unroll
        for (int r = 0; r < 8; ++r) {
            acc00[r] = C[(size_t)(rBase + r)      * N + cCol];
            acc01[r] = C[(size_t)(rBase + r)      * N + cCol + 16];
            acc10[r] = C[(size_t)(rBase + r + 16) * N + cCol];
            acc11[r] = C[(size_t)(rBase + r + 16) * N + cCol + 16];
        }
    } else {
        #pragma unroll
        for (int r = 0; r < 8; ++r) {
            acc00[r] = 0.f; acc01[r] = 0.f; acc10[r] = 0.f; acc11[r] = 0.f;
        }
    }

    const int nch = K / CHUNK;
    // prime the pipeline: chunk 0 -> buffer 0
    tdm_load_tile_bf16(BT + (size_t)n0 * K, 0u,
                       (unsigned)CHUNK, 32u, (unsigned)K, (unsigned)N,
                       (unsigned)K);

    for (int c = 0; c < nch; ++c) {
        if (c + 1 < nch) {
            tdm_load_tile_bf16(BT + (size_t)n0 * K + (size_t)(c + 1) * CHUNK,
                               (unsigned)(((c + 1) & 1) * 32768),
                               (unsigned)CHUNK, 32u, (unsigned)K, (unsigned)N,
                               (unsigned)K);
            __builtin_amdgcn_s_wait_tensorcnt((short)1);  // chunk c landed
        } else {
            __builtin_amdgcn_s_wait_tensorcnt((short)0);
        }
        // TDM wrote Bs behind the optimizer's back: force LDS reloads here.
        asm volatile("" ::: "memory");

        const unsigned short* bs = &Bs[(c & 1) * 8192];
        const int kg0 = c * CHUNK;

        for (int kk = 0; kk < CHUNK; kk += 32) {
            union { u16x8 h[2]; v16bf v; } a0, a1;
            const unsigned short* p0 = A + aRow0 + kg0 + kk + kbaseA;
            a0.h[0] = *(const u16x8*)(p0);
            a0.h[1] = *(const u16x8*)(p0 + 8);
            const unsigned short* p1 = A + aRow1 + kg0 + kk + kbaseA;
            a1.h[0] = *(const u16x8*)(p1);
            a1.h[1] = *(const u16x8*)(p1 + 8);

            union { u16x16 u; v16bf v; } b0, b1;
            b0.u = *(const u16x16*)(bs + bsRow0 + kk + koffB);
            b1.u = *(const u16x16*)(bs + bsRow1 + kk + koffB);

            acc00 = __builtin_amdgcn_wmma_f32_16x16x32_bf16(
                        false, a0.v, false, b0.v, (short)0, acc00, false, false);
            acc01 = __builtin_amdgcn_wmma_f32_16x16x32_bf16(
                        false, a0.v, false, b1.v, (short)0, acc01, false, false);
            acc10 = __builtin_amdgcn_wmma_f32_16x16x32_bf16(
                        false, a1.v, false, b0.v, (short)0, acc10, false, false);
            acc11 = __builtin_amdgcn_wmma_f32_16x16x32_bf16(
                        false, a1.v, false, b1.v, (short)0, acc11, false, false);
        }
    }

    #pragma unroll
    for (int r = 0; r < 8; ++r) {
        C[(size_t)(rBase + r)      * N + cCol]      = acc00[r];
        C[(size_t)(rBase + r)      * N + cCol + 16] = acc01[r];
        C[(size_t)(rBase + r + 16) * N + cCol]      = acc10[r];
        C[(size_t)(rBase + r + 16) * N + cCol + 16] = acc11[r];
    }
}

// ---------------------------------------------------------------------------
// d_inv_sqrt[i] = rsqrt(sum_j adj[i][j] + 1e-8)
// ---------------------------------------------------------------------------
__global__ void __launch_bounds__(256)
degree_kernel(const float* __restrict__ adj, float* __restrict__ dinv, int N)
{
    __shared__ float red[256];
    const int i = blockIdx.x, t = threadIdx.x;
    float s = 0.f;
    for (int j = t; j < N; j += 256) s += adj[(size_t)i * N + j];
    red[t] = s; __syncthreads();
    for (int w = 128; w > 0; w >>= 1) {
        if (t < w) red[t] += red[t + w];
        __syncthreads();
    }
    if (t == 0) dinv[i] = rsqrtf(red[0] + 1e-8f);
}

// b[i][j] = dinv[i] * adj[i][j] * dinv[j]
__global__ void __launch_bounds__(256)
norm_adj_kernel(const float* __restrict__ adj, const float* __restrict__ dinv,
                float* __restrict__ b, int N)
{
    const int i = blockIdx.y;
    const int j = blockIdx.x * 256 + threadIdx.x;
    const size_t idx = (size_t)i * N + j;
    b[idx] = dinv[i] * adj[idx] * dinv[j];
}

// bf16 conversions ----------------------------------------------------------
__global__ void __launch_bounds__(256)
cvt_h_kernel(const float* __restrict__ h, unsigned short* __restrict__ hbf)
{
    const int idx = blockIdx.x * 256 + threadIdx.x;
    hbf[idx] = f32_to_bf16(h[idx]);
}

__global__ void __launch_bounds__(256)
cvt_w_kernel(const float* __restrict__ W, unsigned short* __restrict__ wtbf)
{
    const int idx = blockIdx.x * 256 + threadIdx.x;   // idx over 128*128
    const int k = idx >> 7;          // in_dim index
    const int n = idx & 127;         // out_dim index
    wtbf[n * DIM + k] = f32_to_bf16(W[idx]);          // WT[n][k] = W[k][n]
}

__global__ void __launch_bounds__(256)
cvt_u_kernel(const float* __restrict__ u, unsigned short* __restrict__ ubf,
             unsigned short* __restrict__ utbf)
{
    const int idx = blockIdx.x * 256 + threadIdx.x;   // idx over 4096*128
    const int row = idx >> 7;
    const int col = idx & 127;
    const unsigned short v = f32_to_bf16(u[idx]);
    ubf[idx] = v;                       // U row-major (BT for the V@U^T GEMM)
    utbf[(size_t)col * NTOT + row] = v; // U^T row-major (BT for P@U GEMMs)
}

// hs[i] = h[i,:] . a_src ; hd[i] = h[i,:] . a_dst ---------------------------
__global__ void __launch_bounds__(128)
attn_proj_kernel(const float* __restrict__ h, const float* __restrict__ attn_a,
                 float* __restrict__ hs, float* __restrict__ hd)
{
    __shared__ float red[128];
    const int i = blockIdx.x, t = threadIdx.x;
    const float hv = h[i * DIM + t];

    red[t] = hv * attn_a[t]; __syncthreads();
    for (int w = 64; w > 0; w >>= 1) {
        if (t < w) red[t] += red[t + w];
        __syncthreads();
    }
    if (t == 0) hs[i] = red[0];
    __syncthreads();

    red[t] = hv * attn_a[DIM + t]; __syncthreads();
    for (int w = 64; w > 0; w >>= 1) {
        if (t < w) red[t] += red[t + w];
        __syncthreads();
    }
    if (t == 0) hd[i] = red[0];
}

// Row softmax of b -> bf16 probabilities (row staged in LDS, 3 passes) ------
__global__ void __launch_bounds__(256)
softmax_rows_kernel(const float* __restrict__ Bm, unsigned short* __restrict__ P,
                    int N)
{
    __shared__ float row[NTOT];
    __shared__ float red[256];
    const int i = blockIdx.x, t = threadIdx.x;

    float mx = -3.4e38f;
    for (int j = t; j < N; j += 256) {
        float v = Bm[(size_t)i * N + j];
        row[j] = v; mx = fmaxf(mx, v);
    }
    red[t] = mx; __syncthreads();
    for (int w = 128; w > 0; w >>= 1) {
        if (t < w) red[t] = fmaxf(red[t], red[t + w]);
        __syncthreads();
    }
    mx = red[0]; __syncthreads();

    float sm = 0.f;
    for (int j = t; j < N; j += 256) {
        float e = __expf(row[j] - mx);
        row[j] = e; sm += e;
    }
    red[t] = sm; __syncthreads();
    for (int w = 128; w > 0; w >>= 1) {
        if (t < w) red[t] += red[t + w];
        __syncthreads();
    }
    const float inv = 1.0f / red[0];
    for (int j = t; j < N; j += 256)
        P[(size_t)i * N + j] = f32_to_bf16(row[j] * inv);
}

// Row softmax of leaky_relu(hs[i]+hd[j]) -> bf16 probabilities --------------
__global__ void __launch_bounds__(256)
attn_softmax_kernel(const float* __restrict__ hs, const float* __restrict__ hd,
                    unsigned short* __restrict__ P, int N)
{
    __shared__ float row[NTOT];
    __shared__ float red[256];
    const int i = blockIdx.x, t = threadIdx.x;
    const float hsi = hs[i];

    float mx = -3.4e38f;
    for (int j = t; j < N; j += 256) {
        float x = hsi + hd[j];
        x = (x >= 0.f) ? x : 0.2f * x;      // leaky_relu slope 0.2
        row[j] = x; mx = fmaxf(mx, x);
    }
    red[t] = mx; __syncthreads();
    for (int w = 128; w > 0; w >>= 1) {
        if (t < w) red[t] = fmaxf(red[t], red[t + w]);
        __syncthreads();
    }
    mx = red[0]; __syncthreads();

    float sm = 0.f;
    for (int j = t; j < N; j += 256) {
        float e = __expf(row[j] - mx);
        row[j] = e; sm += e;
    }
    red[t] = sm; __syncthreads();
    for (int w = 128; w > 0; w >>= 1) {
        if (t < w) red[t] += red[t + w];
        __syncthreads();
    }
    const float inv = 1.0f / red[0];
    for (int j = t; j < N; j += 256)
        P[(size_t)i * N + j] = f32_to_bf16(row[j] * inv);
}

// v_bf16 = bf16(squash(s)) per row -----------------------------------------
__global__ void __launch_bounds__(128)
squash_bf_kernel(const float* __restrict__ S, unsigned short* __restrict__ Vbf)
{
    __shared__ float red[128];
    const int i = blockIdx.x, t = threadIdx.x;
    const float x = S[i * DIM + t];
    red[t] = x * x; __syncthreads();
    for (int w = 64; w > 0; w >>= 1) {
        if (t < w) red[t] += red[t + w];
        __syncthreads();
    }
    const float nrm   = sqrtf(red[0]);
    const float scale = nrm / ((1.0f + nrm) * (nrm + 1e-8f));
    Vbf[i * DIM + t] = f32_to_bf16(x * scale);
}

// out = squash(0.6*s + 0.4*s2) per row (f32 output) -------------------------
__global__ void __launch_bounds__(128)
final_squash_kernel(const float* __restrict__ S, const float* __restrict__ S2,
                    float* __restrict__ out)
{
    __shared__ float red[128];
    const int i = blockIdx.x, t = threadIdx.x;
    const float x = 0.6f * S[i * DIM + t] + 0.4f * S2[i * DIM + t];
    red[t] = x * x; __syncthreads();
    for (int w = 64; w > 0; w >>= 1) {
        if (t < w) red[t] += red[t + w];
        __syncthreads();
    }
    const float nrm   = sqrtf(red[0]);
    const float scale = nrm / ((1.0f + nrm) * (nrm + 1e-8f));
    out[i * DIM + t] = x * scale;
}

// ---------------------------------------------------------------------------
// Workspace layout (bytes, all 256B-aligned)
// ---------------------------------------------------------------------------
#define B_OFF     ((size_t)0)                       // 4096*4096*4 = 64 MB f32
#define P_OFF     ((size_t)67108864)                // 4096*4096*2 = 32 MB bf16
#define UHAT_OFF  ((size_t)100663296)               // 4096*128*4
#define UBF_OFF   ((size_t)102760448)               // 4096*128*2
#define UTBF_OFF  ((size_t)103809024)               // 128*4096*2
#define VBF_OFF   ((size_t)104857600)               // 4096*128*2
#define S_OFF     ((size_t)105906176)               // 4096*128*4
#define S2_OFF    ((size_t)108003328)               // 4096*128*4
#define HBF_OFF   ((size_t)110100480)               // 4096*128*2
#define WTBF_OFF  ((size_t)111149056)               // 128*128*2
#define DINV_OFF  ((size_t)111181824)               // 4096*4
#define HS_OFF    ((size_t)111198208)               // 4096*4
#define HD_OFF    ((size_t)111214592)               // 4096*4

extern "C" void kernel_launch(void* const* d_in, const int* in_sizes, int n_in,
                              void* d_out, int out_size, void* d_ws, size_t ws_size,
                              hipStream_t stream)
{
    (void)in_sizes; (void)n_in; (void)out_size; (void)ws_size;

    const float* h    = (const float*)d_in[0];   // (4096,128)
    const float* adj  = (const float*)d_in[1];   // (4096,4096)
    const float* W    = (const float*)d_in[2];   // (128,128)
    const float* attn = (const float*)d_in[3];   // (256,1)
    float* out = (float*)d_out;                  // (4096,128) f32

    char* ws = (char*)d_ws;
    float*          b    = (float*)(ws + B_OFF);
    unsigned short* p    = (unsigned short*)(ws + P_OFF);
    float*          uhat = (float*)(ws + UHAT_OFF);
    unsigned short* ubf  = (unsigned short*)(ws + UBF_OFF);
    unsigned short* utbf = (unsigned short*)(ws + UTBF_OFF);
    unsigned short* vbf  = (unsigned short*)(ws + VBF_OFF);
    float*          s    = (float*)(ws + S_OFF);
    float*          s2   = (float*)(ws + S2_OFF);
    unsigned short* hbf  = (unsigned short*)(ws + HBF_OFF);
    unsigned short* wtbf = (unsigned short*)(ws + WTBF_OFF);
    float*          dinv = (float*)(ws + DINV_OFF);
    float*          hs   = (float*)(ws + HS_OFF);
    float*          hd   = (float*)(ws + HD_OFF);

    const int N = NTOT;
    const size_t GEMM_LDS = 32768;   // 2 x 16KB TDM double buffer

    // 1) symmetric normalization of adj into b
    degree_kernel<<<N, 256, 0, stream>>>(adj, dinv, N);
    norm_adj_kernel<<<dim3(N / 256, N), 256, 0, stream>>>(adj, dinv, b, N);

    // 2) u_hat = h @ W on the WMMA path (bf16 in, f32 out)
    cvt_h_kernel<<<(N * DIM) / 256, 256, 0, stream>>>(h, hbf);
    cvt_w_kernel<<<(DIM * DIM) / 256, 256, 0, stream>>>(W, wtbf);
    wmma_gemm_bf16<<<(N / 32) * (DIM / 32), 32, GEMM_LDS, stream>>>(
        hbf, wtbf, uhat, N, DIM, DIM, 0, 128);
    cvt_u_kernel<<<(N * DIM) / 256, 256, 0, stream>>>(uhat, ubf, utbf);

    // 3) attention projections hs, hd
    attn_proj_kernel<<<N, 128, 0, stream>>>(h, attn, hs, hd);

    // 4) routing iterations
    for (int it = 0; it < 3; ++it) {
        softmax_rows_kernel<<<N, 256, 0, stream>>>(b, p, N);
        // s = softmax(b) @ u_hat
        wmma_gemm_bf16<<<(N / 32) * (DIM / 32), 32, GEMM_LDS, stream>>>(
            p, utbf, s, N, DIM, N, 0, 256);
        if (it < 2) {
            squash_bf_kernel<<<N, 128, 0, stream>>>(s, vbf);
            // b += v @ u_hat^T   (BT = u_hat row-major)
            wmma_gemm_bf16<<<(N / 32) * (N / 32), 32, GEMM_LDS, stream>>>(
                vbf, ubf, b, N, N, DIM, 1, 128);
        }
    }

    // 5) combined @ u_hat = 0.6 * s_last + 0.4 * (softmax(attn) @ u_hat)
    attn_softmax_kernel<<<N, 256, 0, stream>>>(hs, hd, p, N);
    wmma_gemm_bf16<<<(N / 32) * (DIM / 32), 32, GEMM_LDS, stream>>>(
        p, utbf, s2, N, DIM, N, 0, 256);
    final_squash_kernel<<<N, 128, 0, stream>>>(s, s2, out);
}
